// GATLayer_27934467293296
// MI455X (gfx1250) — compile-verified, matrix operand order checked by reading, not run on hardware
//
#include <hip/hip_runtime.h>

#define F_IN      512
#define F_OUT     32
#define HEADS     8
#define C_OUT     256      // HEADS * F_OUT
#define NEG_SLOPE 0.2f
#define EPS_F     1e-16f

typedef __attribute__((ext_vector_type(16))) __bf16 v16bf;
typedef __attribute__((ext_vector_type(8)))  float  v8f;

// ---- order-preserving float <-> uint encoding for atomicMax on f32 ----
__device__ __forceinline__ unsigned enc_f32(float x) {
    unsigned b = __float_as_uint(x);
    return (b & 0x80000000u) ? ~b : (b | 0x80000000u);
}
__device__ __forceinline__ float dec_f32(unsigned k) {
    unsigned b = (k & 0x80000000u) ? (k & 0x7fffffffu) : ~k;
    return __uint_as_float(b);
}

// ------------------------------------------------------------------
// Pack W [H][F_IN][F_OUT] (f32) -> Bpack bf16 in WMMA B-fragment order.
// Logical B^T[n][k] = W[n>>5][k][n&31], n = head*32+f (head-concat cols).
// Fragment order: [nt(16)][kt(16)][lane(32)][16 elems]; per lane L:
//   col n = nt*16 + (L&15), kb = (L>>4)*8,
//   elems 0..7  -> k = kt*32 + kb + e
//   elems 8..15 -> k = kt*32 + 16 + kb + (e-8)
// ------------------------------------------------------------------
__global__ void pack_W(const float* __restrict__ W, __bf16* __restrict__ Bp) {
    int i = blockIdx.x * blockDim.x + threadIdx.x;
    if (i >= F_IN * C_OUT) return;
    int e    = i & 15;
    int lane = (i >> 4) & 31;
    int kt   = (i >> 9) & 15;
    int nt   = i >> 13;
    int n    = nt * 16 + (lane & 15);
    int kb   = (lane >> 4) * 8;
    int k    = kt * 32 + ((e < 8) ? (kb + e) : (16 + kb + (e - 8)));
    int head = n >> 5, fo = n & 31;
    Bp[i] = (__bf16)W[((size_t)head * F_IN + k) * F_OUT + fo];
}

// ------------------------------------------------------------------
// Pack x [N][F_IN] f32 -> Apack bf16 in WMMA A-fragment order:
// [mt][kt(16)][lane(32)][16].  One thread packs one lane-fragment (16 vals).
// ------------------------------------------------------------------
__global__ void pack_X(const float* __restrict__ x, __bf16* __restrict__ Ap, int mtiles) {
    int t = blockIdx.x * blockDim.x + threadIdx.x;
    if (t >= mtiles * 16 * 32) return;
    int lane = t & 31;
    int kt   = (t >> 5) & 15;
    int mt   = t >> 9;
    int row  = mt * 16 + (lane & 15);
    int kb   = (lane >> 4) * 8;
    const float* p = x + (size_t)row * F_IN + kt * 32 + kb;
    float4 f0 = *(const float4*)(p);
    float4 f1 = *(const float4*)(p + 4);
    float4 f2 = *(const float4*)(p + 16);
    float4 f3 = *(const float4*)(p + 20);
    union { v16bf v; __bf16 e[16]; } u;
    u.e[0] = (__bf16)f0.x; u.e[1] = (__bf16)f0.y; u.e[2] = (__bf16)f0.z; u.e[3] = (__bf16)f0.w;
    u.e[4] = (__bf16)f1.x; u.e[5] = (__bf16)f1.y; u.e[6] = (__bf16)f1.z; u.e[7] = (__bf16)f1.w;
    u.e[8] = (__bf16)f2.x; u.e[9] = (__bf16)f2.y; u.e[10]= (__bf16)f2.z; u.e[11]= (__bf16)f2.w;
    u.e[12]= (__bf16)f3.x; u.e[13]= (__bf16)f3.y; u.e[14]= (__bf16)f3.z; u.e[15]= (__bf16)f3.w;
    *(v16bf*)(Ap + (size_t)t * 16) = u.v;
}

// ------------------------------------------------------------------
// GEMM: h[N][256] = A(bf16) x B(bf16), WMMA 16x16x32 bf16, f32 accum.
// WG = 256 threads = 8 waves; wave computes 32x32 tile (2x2 WMMAs).
// Grid: (ceil(mtiles/8), 4); block covers rows bx*128.. cols by*64..
// ------------------------------------------------------------------
__global__ void __launch_bounds__(256) gat_gemm(const __bf16* __restrict__ Ap,
                                                const __bf16* __restrict__ Bp,
                                                float* __restrict__ h, int mtiles) {
    const int lane = threadIdx.x & 31;
    const int w    = threadIdx.x >> 5;
    const int wm   = w & 3, wn = w >> 2;
    const int mt0  = blockIdx.x * 8 + wm * 2;   // wave-uniform guard: EXEC stays all-1s
    if (mt0 >= mtiles) return;
    const int mt1  = mt0 + 1;
    const int nt0  = blockIdx.y * 4 + wn * 2;
    const int nt1  = nt0 + 1;

    const v16bf* A = (const v16bf*)Ap;
    const v16bf* B = (const v16bf*)Bp;
    v8f acc00 = {}, acc01 = {}, acc10 = {}, acc11 = {};

#pragma unroll 4
    for (int kt = 0; kt < F_IN / 32; ++kt) {
        v16bf a0 = A[(size_t)(mt0 * 16 + kt) * 32 + lane];
        v16bf a1 = A[(size_t)(mt1 * 16 + kt) * 32 + lane];
        v16bf b0 = B[(size_t)(nt0 * 16 + kt) * 32 + lane];
        v16bf b1 = B[(size_t)(nt1 * 16 + kt) * 32 + lane];
        acc00 = __builtin_amdgcn_wmma_f32_16x16x32_bf16(false, a0, false, b0, (short)0, acc00, false, false);
        acc01 = __builtin_amdgcn_wmma_f32_16x16x32_bf16(false, a0, false, b1, (short)0, acc01, false, false);
        acc10 = __builtin_amdgcn_wmma_f32_16x16x32_bf16(false, a1, false, b0, (short)0, acc10, false, false);
        acc11 = __builtin_amdgcn_wmma_f32_16x16x32_bf16(false, a1, false, b1, (short)0, acc11, false, false);
    }
    // C/D layout: lane L -> col = L&15; VGPR i -> row = 8*(L>>4) + i
    const int nc0 = nt0 * 16 + (lane & 15);
    const int nc1 = nt1 * 16 + (lane & 15);
    const int mrb = 8 * (lane >> 4);
    float* h0 = h + (size_t)(mt0 * 16 + mrb) * C_OUT;
    float* h1 = h + (size_t)(mt1 * 16 + mrb) * C_OUT;
#pragma unroll
    for (int i = 0; i < 8; ++i) {
        h0[(size_t)i * C_OUT + nc0] = acc00[i];
        h0[(size_t)i * C_OUT + nc1] = acc01[i];
        h1[(size_t)i * C_OUT + nc0] = acc10[i];
        h1[(size_t)i * C_OUT + nc1] = acc11[i];
    }
}

// ------------------------------------------------------------------
// Per-node scores: s_src[n][h] = sum_f h[n][h*32+f]*a_src[h][f]  (one node/block,
// one head per wave, cross-lane reduce).
// ------------------------------------------------------------------
__global__ void node_scores(const float* __restrict__ h, const float* __restrict__ a_src,
                            const float* __restrict__ a_dst, float* __restrict__ s_src,
                            float* __restrict__ s_dst) {
    int n  = blockIdx.x;
    int t  = threadIdx.x;
    int hd = t >> 5, f = t & 31;
    float v  = h[(size_t)n * C_OUT + t];
    float ps = v * a_src[hd * F_OUT + f];
    float pd = v * a_dst[hd * F_OUT + f];
    for (int o = 16; o > 0; o >>= 1) {
        ps += __shfl_xor(ps, o, 32);
        pd += __shfl_xor(pd, o, 32);
    }
    if (f == 0) {
        s_src[n * HEADS + hd] = ps;
        s_dst[n * HEADS + hd] = pd;
    }
}

// out = bias (broadcast), denom = 0, maxkey = sentinel
__global__ void init_out(float* __restrict__ out, const float* __restrict__ bias,
                         float* __restrict__ denom, unsigned* __restrict__ maxkey, int N) {
    int i = blockIdx.x * blockDim.x + threadIdx.x;
    if (i < N * C_OUT)  out[i] = bias[i & (C_OUT - 1)];
    if (i < N * HEADS)  denom[i] = 0.0f;
    if (i == 0)         *maxkey = 0u;
}

// global max of leaky_relu(s_src[src]+s_dst[dst]) over all edges/heads
__global__ void edge_max(const int* __restrict__ ei, const float* __restrict__ s_src,
                         const float* __restrict__ s_dst, unsigned* __restrict__ maxkey, int E) {
    int stride = gridDim.x * blockDim.x;
    float m = -3.402823466e38f;
    for (int e = blockIdx.x * blockDim.x + threadIdx.x; e < E; e += stride) {
        int s = ei[e], d = ei[E + e];
#pragma unroll
        for (int hh = 0; hh < HEADS; ++hh) {
            float v = s_src[s * HEADS + hh] + s_dst[d * HEADS + hh];
            v = (v >= 0.0f) ? v : NEG_SLOPE * v;
            m = fmaxf(m, v);
        }
    }
    for (int o = 16; o > 0; o >>= 1) m = fmaxf(m, __shfl_xor(m, o, 32));
    if ((threadIdx.x & 31) == 0) atomicMax(maxkey, enc_f32(m));
}

// denom[dst][h] += exp(e - m)
__global__ void edge_denom(const int* __restrict__ ei, const float* __restrict__ s_src,
                           const float* __restrict__ s_dst, const unsigned* __restrict__ maxkey,
                           float* __restrict__ denom, int E) {
    int e = blockIdx.x * blockDim.x + threadIdx.x;
    if (e >= E) return;
    float m = dec_f32(*maxkey);
    int s = ei[e], d = ei[E + e];
#pragma unroll
    for (int hh = 0; hh < HEADS; ++hh) {
        float v = s_src[s * HEADS + hh] + s_dst[d * HEADS + hh];
        v = (v >= 0.0f) ? v : NEG_SLOPE * v;
        unsafeAtomicAdd(&denom[d * HEADS + hh], __expf(v - m));
    }
}

// one wave per edge: out[dst] += alpha * h[src]   (lane covers 8 contiguous f32)
__global__ void __launch_bounds__(256) edge_aggregate(
        const int* __restrict__ ei, const float* __restrict__ s_src,
        const float* __restrict__ s_dst, const float* __restrict__ denom,
        const unsigned* __restrict__ maxkey, const float* __restrict__ h,
        float* __restrict__ out, int E) {
    int lane = threadIdx.x & 31;
    int e = blockIdx.x * 8 + (threadIdx.x >> 5);
    if (e >= E) return;
    float m = dec_f32(*maxkey);
    int s = ei[e], d = ei[E + e];
    int head = lane >> 2;                      // lane*8 / 32
    float v = s_src[s * HEADS + head] + s_dst[d * HEADS + head];
    v = (v >= 0.0f) ? v : NEG_SLOPE * v;
    float alpha = __expf(v - m) / (denom[d * HEADS + head] + EPS_F);
    const float4* hp = (const float4*)(h + (size_t)s * C_OUT + lane * 8);
    float4 h0 = hp[0], h1 = hp[1];
    float* op = out + (size_t)d * C_OUT + lane * 8;
    unsafeAtomicAdd(op + 0, alpha * h0.x);
    unsafeAtomicAdd(op + 1, alpha * h0.y);
    unsafeAtomicAdd(op + 2, alpha * h0.z);
    unsafeAtomicAdd(op + 3, alpha * h0.w);
    unsafeAtomicAdd(op + 4, alpha * h1.x);
    unsafeAtomicAdd(op + 5, alpha * h1.y);
    unsafeAtomicAdd(op + 6, alpha * h1.z);
    unsafeAtomicAdd(op + 7, alpha * h1.w);
}

extern "C" void kernel_launch(void* const* d_in, const int* in_sizes, int n_in,
                              void* d_out, int out_size, void* d_ws, size_t ws_size,
                              hipStream_t stream) {
    const float* x     = (const float*)d_in[0];
    const int*   ei    = (const int*)d_in[1];
    const float* W     = (const float*)d_in[2];
    const float* a_src = (const float*)d_in[3];
    const float* a_dst = (const float*)d_in[4];
    const float* bias  = (const float*)d_in[5];
    float* out = (float*)d_out;

    const int N = in_sizes[0] / F_IN;       // 40000
    const int E = in_sizes[1] / 2;          // 640000
    const int mtiles = N / 16;              // 2500

    // workspace carve-up (all offsets 256B-aligned)
    char* ws = (char*)d_ws;
    size_t off = 0;
    __bf16* Bp = (__bf16*)(ws + off);  off += (size_t)F_IN * C_OUT * 2;          // 256 KB
    off = (off + 255) & ~(size_t)255;
    __bf16* Ap = (__bf16*)(ws + off);  off += (size_t)N * F_IN * 2;              // 40 MB
    off = (off + 255) & ~(size_t)255;
    float* h     = (float*)(ws + off); off += (size_t)N * C_OUT * 4;             // 40 MB
    off = (off + 255) & ~(size_t)255;
    float* s_src = (float*)(ws + off); off += (size_t)N * HEADS * 4;
    off = (off + 255) & ~(size_t)255;
    float* s_dst = (float*)(ws + off); off += (size_t)N * HEADS * 4;
    off = (off + 255) & ~(size_t)255;
    float* denom = (float*)(ws + off); off += (size_t)N * HEADS * 4;
    off = (off + 255) & ~(size_t)255;
    unsigned* maxkey = (unsigned*)(ws + off);

    // 1) pack weights + activations to bf16 WMMA fragment order
    pack_W<<<(F_IN * C_OUT + 255) / 256, 256, 0, stream>>>(W, Bp);
    pack_X<<<(mtiles * 16 * 32 + 255) / 256, 256, 0, stream>>>(x, Ap, mtiles);

    // 2) h = x @ W  (WMMA bf16 -> f32)
    dim3 ggrid((mtiles + 7) / 8, C_OUT / 64);
    gat_gemm<<<ggrid, 256, 0, stream>>>(Ap, Bp, h, mtiles);

    // 3) per-node attention scores
    node_scores<<<N, 256, 0, stream>>>(h, a_src, a_dst, s_src, s_dst);

    // 4) init out=bias, denom=0, maxkey=sentinel
    init_out<<<(N * C_OUT + 255) / 256, 256, 0, stream>>>(out, bias, denom, maxkey, N);

    // 5) global max shift (matches reference e - max(e))
    edge_max<<<2048, 256, 0, stream>>>(ei, s_src, s_dst, maxkey, E);

    // 6) softmax denominators per (dst, head)
    edge_denom<<<(E + 255) / 256, 256, 0, stream>>>(ei, s_src, s_dst, maxkey, denom, E);

    // 7) weighted scatter-add of messages (one wave per edge)
    edge_aggregate<<<(E + 7) / 8, 256, 0, stream>>>(ei, s_src, s_dst, denom, maxkey, h, out, E);
}
